// AttackHead_59605556133929
// MI455X (gfx1250) — matrix operand order repaired and checked by reading, not compiled, not run on hardware
//
#include <hip/hip_runtime.h>

typedef __attribute__((ext_vector_type(16))) __bf16 v16bf;
typedef __attribute__((ext_vector_type(8)))  __bf16 v8bf;
typedef __attribute__((ext_vector_type(8)))  float  v8f;

#define B_ 64
#define N_ 4096
#define E_ 8192
#define D_ 256
#define H_ 256
#define M_ 64
#define ROWS_ 64   // edges per block (4 row-tiles of 16)

static __device__ __forceinline__ v8f wmma_bf16(v16bf a, v16bf b, v8f c) {
  // (neg_a, A, neg_b, B, c_mod, C, reuse_a, reuse_b)
  return __builtin_amdgcn_wmma_f32_16x16x32_bf16(false, a, false, b, (short)0, c, false, false);
}

static __device__ __forceinline__ v8f splat8(float x) {
  v8f r;
#pragma unroll
  for (int i = 0; i < 8; ++i) r[i] = x;
  return r;
}

// Load WMMA 16-bit A fragment from a bf16 row in LDS (row-major, this lane's row).
// Lane (r, half) holds K = ktbase + 16*half + [0..15] in elems 0..15
// (contiguous order, matching the packed B layout).
static __device__ __forceinline__ v16bf load_a(const __bf16* rowptr, int kt, int half) {
  const __bf16* p = rowptr + kt * 32 + 16 * half;
  v8bf lo = *(const v8bf*)p;
  v8bf hi = *(const v8bf*)(p + 8);
  return __builtin_shufflevector(lo, hi, 0, 1, 2, 3, 4, 5, 6, 7, 8, 9, 10, 11, 12, 13, 14, 15);
}

static __device__ __forceinline__ void store_relu_bf16(const v8f& acc, __bf16* dst, int stride,
                                                       int roff, int col, int half) {
#pragma unroll
  for (int v = 0; v < 8; ++v)
    dst[(roff + v + 8 * half) * stride + col] = (__bf16)fmaxf(acc[v], 0.0f);
}

// Re-pack a K x Nn row-major f32 weight matrix into bf16 WMMA-B lane order:
// dst[((kt*Nn + n)*2 + half)*16 + j] = W[(kt*32 + half*16 + j)*Nn + n]
__global__ void pack_b_kernel(const float* __restrict__ W, __bf16* __restrict__ dst,
                              int Nn, int total) {
  int idx = blockIdx.x * blockDim.x + threadIdx.x;
  if (idx >= total) return;
  int kt   = idx / (Nn * 32);
  int rem  = idx - kt * Nn * 32;
  int n    = rem >> 5;
  int l    = rem & 31;
  int half = l >> 4;
  int j    = l & 15;
  int k    = kt * 32 + half * 16 + j;
  dst[idx] = (__bf16)W[k * Nn + n];
}

__global__ __launch_bounds__(256) void attackhead_fused(
    const float* __restrict__ emb, const float* __restrict__ army,
    const int2* __restrict__ edges,
    const float* __restrict__ b1, const float* __restrict__ W2,
    const float* __restrict__ b2, const float* __restrict__ a1,
    const float* __restrict__ a2,
    const __bf16* __restrict__ W1p, const __bf16* __restrict__ A1p,
    const __bf16* __restrict__ A2p,
    float* __restrict__ edge_out, float* __restrict__ army_out) {
  // 64 edges per block (4 row-tiles of 16). Padded row strides rotate LDS banks.
  __shared__ __align__(16) __bf16 sX[ROWS_ * 520];  // x = [src_e | tgt_e], bf16
  __shared__ __align__(16) __bf16 sH[ROWS_ * 264];  // relu(x@W1+b1)
  __shared__ __align__(16) __bf16 sG[ROWS_ * 136];  // relu(x@A1+a1)
  __shared__ float sMax[ROWS_];
  __shared__ float sPen[ROWS_];

  const int t = threadIdx.x;

  // ---- Stage 1: gather endpoint embeddings -> LDS (bf16), compute penalties ----
  {
    const int row = t >> 2, sub = t & 3;          // 4 threads per row, 128 cols each
    const int eg = blockIdx.x * ROWS_ + row;
    const int b = eg >> 13;                        // E_ == 8192
    const int e = eg & (E_ - 1);
    int2 se = edges[b * E_ + e];
    int src = se.x, tgt = se.y;
    int sc = src < 0 ? 0 : (src > N_ - 1 ? N_ - 1 : src);
    int tc = tgt < 0 ? 0 : (tgt > N_ - 1 ? N_ - 1 : tgt);
    if (sub == 0) {
      float sa = army[b * N_ + sc];
      float ta = army[b * N_ + tc];
      sMax[row] = sa - 1.0f;
      bool valid = (src >= 0) && (tgt >= 0);
      float pen = 0.0f;
      if (valid && ((sa <= 2.0f) || (ta >= 3.0f * sa))) pen += 1.0f;
      if (valid && (src == tgt)) pen += 100.0f;
      sPen[row] = pen;
    }
    const int node = (sub < 2) ? sc : tc;
    const float* sp = emb + ((long)b * N_ + node) * D_ + (sub & 1) * 128;
    __bf16* op = &sX[row * 520 + sub * 128];
#pragma unroll 8
    for (int c = 0; c < 128; c += 4) {
      float4 v = *(const float4*)(sp + c);
      op[c + 0] = (__bf16)v.x; op[c + 1] = (__bf16)v.y;
      op[c + 2] = (__bf16)v.z; op[c + 3] = (__bf16)v.w;
    }
  }
  __syncthreads();

  const int lane = t & 31;
  const int w    = t >> 5;        // 8 waves
  const int nl   = lane & 15;
  const int half = lane >> 4;

  // ---- Stage 2: layer-1 GEMMs via WMMA bf16 (K = 512 -> 16 k-tiles) ----
  // Each wave: 3 column tiles x 4 row tiles; each B fragment feeds 4 WMMAs.
  {
    const float bias0 = b1[w * 16 + nl];
    const float bias1 = b1[w * 16 + 128 + nl];
    const float bias2v = a1[w * 16 + nl];
    v8f acc[4][3];
#pragma unroll
    for (int rt = 0; rt < 4; ++rt) {
      acc[rt][0] = splat8(bias0);
      acc[rt][1] = splat8(bias1);
      acc[rt][2] = splat8(bias2v);
    }
    const __bf16* bp0 = W1p + ((w * 16 + nl) * 32 + half * 16);
    const __bf16* bp1 = W1p + ((w * 16 + 128 + nl) * 32 + half * 16);
    const __bf16* bp2 = A1p + ((w * 16 + nl) * 32 + half * 16);
    const __bf16* xrow[4];
#pragma unroll
    for (int rt = 0; rt < 4; ++rt) xrow[rt] = &sX[(rt * 16 + nl) * 520];

#pragma unroll 2
    for (int kt = 0; kt < 16; ++kt) {
      v16bf a[4];
#pragma unroll
      for (int rt = 0; rt < 4; ++rt) a[rt] = load_a(xrow[rt], kt, half);
      v16bf bw0 = *(const v16bf*)(bp0 + kt * (256 * 32));
#pragma unroll
      for (int rt = 0; rt < 4; ++rt) acc[rt][0] = wmma_bf16(a[rt], bw0, acc[rt][0]);
      v16bf bw1 = *(const v16bf*)(bp1 + kt * (256 * 32));
#pragma unroll
      for (int rt = 0; rt < 4; ++rt) acc[rt][1] = wmma_bf16(a[rt], bw1, acc[rt][1]);
      v16bf bw2 = *(const v16bf*)(bp2 + kt * (128 * 32));
#pragma unroll
      for (int rt = 0; rt < 4; ++rt) acc[rt][2] = wmma_bf16(a[rt], bw2, acc[rt][2]);
    }
#pragma unroll
    for (int rt = 0; rt < 4; ++rt) {
      store_relu_bf16(acc[rt][0], sH, 264, rt * 16, w * 16 + nl,       half);
      store_relu_bf16(acc[rt][1], sH, 264, rt * 16, w * 16 + 128 + nl, half);
      store_relu_bf16(acc[rt][2], sG, 136, rt * 16, w * 16 + nl,       half);
    }
  }
  __syncthreads();

  // ---- Stage 3 ----
  if (w < 4) {
    // Army head: g[64x128] @ A2[128x64] (+a2), mask, store. 4 row-tiles per wave.
    const int m = w * 16 + nl;
    const float mf = (float)m;
    const float bias = a2[m];
    const __bf16* bp = A2p + (m * 32 + half * 16);
#pragma unroll
    for (int rt = 0; rt < 4; ++rt) {
      v8f acc = splat8(bias);
      const __bf16* grow = &sG[(rt * 16 + nl) * 136];
#pragma unroll
      for (int kt = 0; kt < 4; ++kt) {
        v16bf av = load_a(grow, kt, half);
        v16bf bv = *(const v16bf*)(bp + kt * (64 * 32));
        acc = wmma_bf16(av, bv, acc);
      }
#pragma unroll
      for (int v = 0; v < 8; ++v) {
        int row = rt * 16 + v + 8 * half;
        float val = (mf <= sMax[row]) ? acc[v] : -1.0e9f;
        army_out[((long)blockIdx.x * ROWS_ + row) * M_ + m] = val;
      }
    }
  } else {
    // Edge head: h[row] . W2 + b2 - penalties; 16 rows per wave. W2 hoisted.
    const float bias2 = b2[0];
    float w2r[8];
#pragma unroll
    for (int j = 0; j < 8; ++j) w2r[j] = W2[lane * 8 + j];
    for (int rr = 0; rr < 16; ++rr) {
      const int row = (w - 4) * 16 + rr;
      v8bf hv = *(const v8bf*)&sH[row * 264 + lane * 8];
      float s = 0.0f;
#pragma unroll
      for (int j = 0; j < 8; ++j) s += (float)hv[j] * w2r[j];
#pragma unroll
      for (int off = 16; off > 0; off >>= 1) s += __shfl_xor(s, off, 32);
      if (lane == 0) edge_out[blockIdx.x * ROWS_ + row] = s + bias2 - sPen[row];
    }
  }
}

extern "C" void kernel_launch(void* const* d_in, const int* in_sizes, int n_in,
                              void* d_out, int out_size, void* d_ws, size_t ws_size,
                              hipStream_t stream) {
  const float* emb   = (const float*)d_in[0];
  const float* army  = (const float*)d_in[1];
  const int2*  edges = (const int2*)d_in[2];
  const float* W1    = (const float*)d_in[3];
  const float* b1    = (const float*)d_in[4];
  const float* W2    = (const float*)d_in[5];
  const float* b2    = (const float*)d_in[6];
  const float* A1    = (const float*)d_in[7];
  const float* a1    = (const float*)d_in[8];
  const float* A2    = (const float*)d_in[9];
  const float* a2    = (const float*)d_in[10];

  // Workspace: bf16 WMMA-B packed weights (400 KB total, L2-resident).
  __bf16* W1p = (__bf16*)d_ws;          // 512*256 elems
  __bf16* A1p = W1p + 512 * 256;        // 512*128 elems
  __bf16* A2p = A1p + 512 * 128;        // 128*64 elems

  pack_b_kernel<<<(512 * 256 + 255) / 256, 256, 0, stream>>>(W1, W1p, 256, 512 * 256);
  pack_b_kernel<<<(512 * 128 + 255) / 256, 256, 0, stream>>>(A1, A1p, 128, 512 * 128);
  pack_b_kernel<<<(128 * 64 + 255) / 256, 256, 0, stream>>>(A2, A2p, 64, 128 * 64);

  float* edge_out = (float*)d_out;
  float* army_out = edge_out + (long)B_ * E_;

  attackhead_fused<<<(B_ * E_) / ROWS_, 256, 0, stream>>>(
      emb, army, edges, b1, W2, b2, a1, a2, W1p, A1p, A2p, edge_out, army_out);
}
